// ExphormerModel_41798621725212
// MI455X (gfx1250) — matrix operand order repaired and checked by reading, not compiled, not run on hardware
//
#include <hip/hip_runtime.h>
#include <hip/hip_bf16.h>

// ---------------------------------------------------------------------------
// CDNA5 (gfx1250, wave32) implementation of the Exphormer-style GraphSAGE ref.
// All GEMMs run on v_wmma_f32_16x16x32_bf16 (bf16 in, fp32 accumulate).
// Feature matrices live in bf16 (90 MB -> L2-resident on the 192 MB L2).
// ---------------------------------------------------------------------------

typedef __attribute__((ext_vector_type(16))) __bf16 v16bf;
typedef __attribute__((ext_vector_type(8)))  __bf16 v8bf;
typedef __attribute__((ext_vector_type(4)))  __bf16 v4bf;
typedef __attribute__((ext_vector_type(8)))  float  v8f;

// Problem constants (from the reference)
static constexpr int HD   = 128;
static constexpr int NPLc = 100000;
static constexpr int NTRc = 200000;
static constexpr int NARc = 50000;
static constexpr int NTOT = NPLc + NTRc + NARc;   // 350000, divisible by 16
static constexpr int OFFTR = NPLc;
static constexpr int OFFAR = NPLc + NTRc;
static constexpr long long EPTc = 500000;
static constexpr long long ETAc = 200000;
static constexpr long long ETOT = 2 * EPTc + 2 * ETAc;  // 1.4M directed edges
static constexpr int Lc   = 100000;

// ---------------------------------------------------------------------------
// small helpers
// ---------------------------------------------------------------------------
__global__ void zero_f32(float* __restrict__ p, long long n) {
  long long i = (long long)blockIdx.x * blockDim.x + threadIdx.x;
  long long stride = (long long)gridDim.x * blockDim.x;
  for (; i < n; i += stride) p[i] = 0.0f;
}

__global__ void cvt_f32_bf16(const float* __restrict__ s, __bf16* __restrict__ d, int n) {
  int i = blockIdx.x * blockDim.x + threadIdx.x;
  if (i < n) d[i] = (__bf16)s[i];
}

// encode playlist + artist nodes: table gather + type embedding, store bf16
__global__ void encode_gather(const int* __restrict__ pl_ids, const float* __restrict__ pl_table,
                              const int* __restrict__ ar_ids, const float* __restrict__ ar_table,
                              const float* __restrict__ type_table, __bf16* __restrict__ x) {
  long long t = (long long)blockIdx.x * blockDim.x + threadIdx.x;
  long long total = (long long)(NPLc + NARc) * HD;
  if (t >= total) return;
  int c = (int)(t & (HD - 1));
  long long r = t >> 7;
  if (r < NPLc) {
    float v = pl_table[(size_t)pl_ids[r] * HD + c] + type_table[c];
    x[(size_t)r * HD + c] = (__bf16)v;
  } else {
    long long a = r - NPLc;
    float v = ar_table[(size_t)ar_ids[a] * HD + c] + type_table[2 * HD + c];
    x[((size_t)OFFAR + a) * HD + c] = (__bf16)v;
  }
}

// ---------------------------------------------------------------------------
// WMMA GEMM: track encoder  x_tr = track_x @ Wtr^T + btr + type[1]
// one wave -> one 16x16 output tile; K=128 via 4 chunks of 32
// ---------------------------------------------------------------------------
__global__ __launch_bounds__(256) void track_gemm(const float* __restrict__ tx,
                                                  const __bf16* __restrict__ Wb,
                                                  const float* __restrict__ btr,
                                                  const float* __restrict__ ttab,
                                                  __bf16* __restrict__ x) {
  const int lane = threadIdx.x & 31;
  const int tile = blockIdx.x * 8 + (threadIdx.x >> 5);
  const int RB = NTRc / 16;                 // 12500
  const int rb = tile >> 3;
  if (rb >= RB) return;
  const int ocb = (tile & 7) << 4;          // output-column base
  const int n = lane & 15;                  // A row within tile / B column
  const int half = lane >> 4;
  const int rowBase = rb << 4;
  const float bias = btr[ocb + n] + ttab[HD + ocb + n];
  v8f c;
#pragma unroll
  for (int i = 0; i < 8; ++i) c[i] = bias;

  const float*  arow = tx + (size_t)(rowBase + n) * HD;
  const __bf16* brow = Wb + (size_t)(ocb + n) * HD;
#pragma unroll
  for (int kc = 0; kc < 4; ++kc) {
    const int ab = kc * 32 + half * 16;     // A chunk: {ab..ab+15}
    v8f lo = *(const v8f*)(arow + ab);
    v8f hi = *(const v8f*)(arow + ab + 8);
    v16bf A;
#pragma unroll
    for (int i = 0; i < 8; ++i) { A[i] = (__bf16)lo[i]; A[8 + i] = (__bf16)hi[i]; }
    v16bf B = *(const v16bf*)(brow + kc * 32 + half * 16);
    c = __builtin_amdgcn_wmma_f32_16x16x32_bf16(false, A, false, B, (short)0, c, false, false);
  }
#pragma unroll
  for (int i = 0; i < 8; ++i) {
    int row = rowBase + i + (half << 3);
    x[((size_t)OFFTR + row) * HD + ocb + n] = (__bf16)c[i];
  }
}

// ---------------------------------------------------------------------------
// homogeneous edge list (forward + reverse)
// ---------------------------------------------------------------------------
__device__ __forceinline__ void edge_sd(long long e,
                                        const int* __restrict__ src_pt, const int* __restrict__ dst_pt,
                                        const int* __restrict__ src_ta, const int* __restrict__ dst_ta,
                                        int& s, int& d) {
  if (e < EPTc)                 { s = src_pt[e];                        d = dst_pt[e] + OFFTR; }
  else if (e < 2 * EPTc)        { long long i = e - EPTc;       s = dst_pt[i] + OFFTR; d = src_pt[i]; }
  else if (e < 2 * EPTc + ETAc) { long long i = e - 2 * EPTc;   s = src_ta[i] + OFFTR; d = dst_ta[i] + OFFAR; }
  else                          { long long i = e - 2 * EPTc - ETAc; s = dst_ta[i] + OFFAR; d = src_ta[i] + OFFTR; }
}

__global__ void degree_kernel(const int* __restrict__ src_pt, const int* __restrict__ dst_pt,
                              const int* __restrict__ src_ta, const int* __restrict__ dst_ta,
                              float* __restrict__ deg) {
  long long e = (long long)blockIdx.x * blockDim.x + threadIdx.x;
  if (e >= ETOT) return;
  int s, d;
  edge_sd(e, src_pt, dst_pt, src_ta, dst_ta, s, d);
  atomicAdd(&deg[d], 1.0f);
}

__global__ void invdeg_kernel(float* __restrict__ deg) {
  int i = blockIdx.x * blockDim.x + threadIdx.x;
  if (i < NTOT) deg[i] = 1.0f / fmaxf(deg[i], 1.0f);
}

// scatter-add of source features into fp32 agg: one wave per edge (4 ch/lane)
__global__ __launch_bounds__(256) void scatter_kernel(const __bf16* __restrict__ x,
                                                      float* __restrict__ agg,
                                                      const int* __restrict__ src_pt, const int* __restrict__ dst_pt,
                                                      const int* __restrict__ src_ta, const int* __restrict__ dst_ta) {
  const int lane = threadIdx.x & 31;
  long long e = (long long)blockIdx.x * 8 + (threadIdx.x >> 5);
  if (e >= ETOT) return;
  int s, d;
  edge_sd(e, src_pt, dst_pt, src_ta, dst_ta, s, d);
  v4bf v = *(const v4bf*)(x + (size_t)s * HD + lane * 4);
  float* ap = agg + (size_t)d * HD + lane * 4;
#pragma unroll
  for (int j = 0; j < 4; ++j) atomicAdd(ap + j, (float)v[j]);
}

// ---------------------------------------------------------------------------
// SAGE layer: out = relu( (agg*inv_deg) @ Wl^T + bl + x @ Wr^T ), bf16 out
// one wave -> 16x16 tile; 2 WMMA per K-chunk (8 wmma per tile)
// ---------------------------------------------------------------------------
__global__ __launch_bounds__(256) void sage_kernel(const __bf16* __restrict__ xin,
                                                   const float* __restrict__ agg,
                                                   const float* __restrict__ invdeg,
                                                   const __bf16* __restrict__ Wl,
                                                   const __bf16* __restrict__ Wr,
                                                   const float* __restrict__ bl,
                                                   __bf16* __restrict__ xout) {
  const int lane = threadIdx.x & 31;
  const int tile = blockIdx.x * 8 + (threadIdx.x >> 5);
  const int RB = NTOT / 16;                 // 21875
  const int rb = tile >> 3;
  if (rb >= RB) return;
  const int ocb = (tile & 7) << 4;
  const int n = lane & 15;
  const int half = lane >> 4;
  const int rowBase = rb << 4;
  const int aRow = rowBase + n;
  const float scale = invdeg[aRow];
  const float bias = bl[ocb + n];
  v8f c;
#pragma unroll
  for (int i = 0; i < 8; ++i) c[i] = bias;

  const __bf16* xrow = xin + (size_t)aRow * HD;
  const float*  grow = agg + (size_t)aRow * HD;
  const __bf16* wl   = Wl + (size_t)(ocb + n) * HD;
  const __bf16* wr   = Wr + (size_t)(ocb + n) * HD;
#pragma unroll
  for (int kc = 0; kc < 4; ++kc) {
    const int ab = kc * 32 + half * 16;
    // root path (bf16 direct)
    v8bf xlo = *(const v8bf*)(xrow + ab);
    v8bf xhi = *(const v8bf*)(xrow + ab + 8);
    v16bf Ax;
#pragma unroll
    for (int i = 0; i < 8; ++i) { Ax[i] = xlo[i]; Ax[8 + i] = xhi[i]; }
    // aggregated path (fp32, scale by inv_deg, cvt bf16)
    v8f glo = *(const v8f*)(grow + ab);
    v8f ghi = *(const v8f*)(grow + ab + 8);
    v16bf Ag;
#pragma unroll
    for (int i = 0; i < 8; ++i) {
      Ag[i]     = (__bf16)(glo[i] * scale);
      Ag[8 + i] = (__bf16)(ghi[i] * scale);
    }
    v16bf Bl = *(const v16bf*)(wl + kc * 32 + half * 16);
    v16bf Br = *(const v16bf*)(wr + kc * 32 + half * 16);
    c = __builtin_amdgcn_wmma_f32_16x16x32_bf16(false, Ax, false, Br, (short)0, c, false, false);
    c = __builtin_amdgcn_wmma_f32_16x16x32_bf16(false, Ag, false, Bl, (short)0, c, false, false);
  }
#pragma unroll
  for (int i = 0; i < 8; ++i) {
    int row = rowBase + i + (half << 3);
    xout[(size_t)row * HD + ocb + n] = (__bf16)fmaxf(c[i], 0.0f);
  }
}

// ---------------------------------------------------------------------------
// prediction head part 1: h = relu( concat(x[pl], x[tr]) @ Wp1^T + bp1 )
// K = 256 (8 chunks of 32); per-lane A chunks gather from pl or tr row
// ---------------------------------------------------------------------------
__global__ __launch_bounds__(256) void predh_kernel(const __bf16* __restrict__ x,
                                                    const int* __restrict__ lab_row,
                                                    const int* __restrict__ lab_col,
                                                    const __bf16* __restrict__ Wp1b,
                                                    const float* __restrict__ bp1,
                                                    __bf16* __restrict__ h) {
  const int lane = threadIdx.x & 31;
  const int tile = blockIdx.x * 8 + (threadIdx.x >> 5);
  const int RB = Lc / 16;                   // 6250
  const int rb = tile >> 3;
  if (rb >= RB) return;
  const int ocb = (tile & 7) << 4;
  const int n = lane & 15;
  const int half = lane >> 4;
  const int rowBase = rb << 4;
  const int lrow = rowBase + n;
  const size_t pl = (size_t)lab_row[lrow] * HD;
  const size_t tr = ((size_t)OFFTR + lab_col[lrow]) * HD;
  const float bias = bp1[ocb + n];
  v8f c;
#pragma unroll
  for (int i = 0; i < 8; ++i) c[i] = bias;

  const __bf16* brow = Wp1b + (size_t)(ocb + n) * (2 * HD);
#pragma unroll
  for (int kc = 0; kc < 8; ++kc) {
    const int a0 = kc * 32 + half * 16;     // 16-aligned -> never crosses 128
    const int a1 = a0 + 8;
    const __bf16* p0 = x + (a0 < HD ? pl + a0 : tr + (a0 - HD));
    const __bf16* p1 = x + (a1 < HD ? pl + a1 : tr + (a1 - HD));
    v8bf lo = *(const v8bf*)p0;
    v8bf hi = *(const v8bf*)p1;
    v16bf A;
#pragma unroll
    for (int i = 0; i < 8; ++i) { A[i] = lo[i]; A[8 + i] = hi[i]; }
    v16bf B = *(const v16bf*)(brow + kc * 32 + half * 16);
    c = __builtin_amdgcn_wmma_f32_16x16x32_bf16(false, A, false, B, (short)0, c, false, false);
  }
#pragma unroll
  for (int i = 0; i < 8; ++i) {
    int row = rowBase + i + (half << 3);
    h[(size_t)row * HD + ocb + n] = (__bf16)fmaxf(c[i], 0.0f);
  }
}

// prediction head part 2: logits = h @ Wp2^T + bp2 (wave-per-label reduction)
__global__ __launch_bounds__(256) void logits_kernel(const __bf16* __restrict__ h,
                                                     const float* __restrict__ Wp2,
                                                     const float* __restrict__ bp2,
                                                     float* __restrict__ out) {
  const int lane = threadIdx.x & 31;
  long long l = (long long)blockIdx.x * 8 + (threadIdx.x >> 5);
  if (l >= Lc) return;
  v4bf hv = *(const v4bf*)(h + (size_t)l * HD + lane * 4);
  float s = 0.0f;
#pragma unroll
  for (int j = 0; j < 4; ++j) s += (float)hv[j] * Wp2[lane * 4 + j];
#pragma unroll
  for (int o = 16; o > 0; o >>= 1) s += __shfl_down(s, o, 32);
  if (lane == 0) out[l] = s + bp2[0];
}

// ---------------------------------------------------------------------------
// host launcher
// ---------------------------------------------------------------------------
extern "C" void kernel_launch(void* const* d_in, const int* in_sizes, int n_in,
                              void* d_out, int out_size, void* d_ws, size_t ws_size,
                              hipStream_t stream) {
  const int*   pl_ids   = (const int*)  d_in[0];
  const int*   ar_ids   = (const int*)  d_in[1];
  const float* track_x  = (const float*)d_in[2];
  const int*   src_pt   = (const int*)  d_in[3];
  const int*   dst_pt   = (const int*)  d_in[4];
  const int*   src_ta   = (const int*)  d_in[5];
  const int*   dst_ta   = (const int*)  d_in[6];
  const int*   lab_row  = (const int*)  d_in[7];
  const int*   lab_col  = (const int*)  d_in[8];
  const float* pl_table = (const float*)d_in[9];
  const float* ar_table = (const float*)d_in[10];
  const float* Wtr      = (const float*)d_in[11];
  const float* btr      = (const float*)d_in[12];
  const float* ttab     = (const float*)d_in[13];
  const float* Wl0      = (const float*)d_in[14];
  const float* bl0      = (const float*)d_in[15];
  const float* Wr0      = (const float*)d_in[16];
  const float* Wl1      = (const float*)d_in[17];
  const float* bl1      = (const float*)d_in[18];
  const float* Wr1      = (const float*)d_in[19];
  const float* Wp1      = (const float*)d_in[20];
  const float* bp1      = (const float*)d_in[21];
  const float* Wp2      = (const float*)d_in[22];
  const float* bp2      = (const float*)d_in[23];
  float* out = (float*)d_out;

  // workspace layout (bytes, 256-aligned)
  char* ws = (char*)d_ws;
  __bf16* Wtr_b = (__bf16*)(ws + 0);                 //  32768 B
  __bf16* Wl0_b = (__bf16*)(ws + 32768);             //  32768 B
  __bf16* Wr0_b = (__bf16*)(ws + 65536);             //  32768 B
  __bf16* Wl1_b = (__bf16*)(ws + 98304);             //  32768 B
  __bf16* Wr1_b = (__bf16*)(ws + 131072);            //  32768 B
  __bf16* Wp1_b = (__bf16*)(ws + 163840);            //  65536 B
  float*  deg   = (float*) (ws + 229376);            //  1.4 MB
  float*  agg   = (float*) (ws + 1629440);           //  179.2 MB
  __bf16* xA    = (__bf16*)(ws + 180829440ULL);      //  89.6 MB
  __bf16* xB    = (__bf16*)(ws + 270429440ULL);      //  89.6 MB
  __bf16* hbuf  = (__bf16*)(ws + 360029440ULL);      //  25.6 MB

  const int B256 = 256;

  // 1) convert weights to bf16
  cvt_f32_bf16<<<(16384 + 255) / 256, B256, 0, stream>>>(Wtr, Wtr_b, 16384);
  cvt_f32_bf16<<<(16384 + 255) / 256, B256, 0, stream>>>(Wl0, Wl0_b, 16384);
  cvt_f32_bf16<<<(16384 + 255) / 256, B256, 0, stream>>>(Wr0, Wr0_b, 16384);
  cvt_f32_bf16<<<(16384 + 255) / 256, B256, 0, stream>>>(Wl1, Wl1_b, 16384);
  cvt_f32_bf16<<<(16384 + 255) / 256, B256, 0, stream>>>(Wr1, Wr1_b, 16384);
  cvt_f32_bf16<<<(32768 + 255) / 256, B256, 0, stream>>>(Wp1, Wp1_b, 32768);

  // 2) encode nodes into xA (bf16)
  {
    long long total = (long long)(NPLc + NARc) * HD;
    encode_gather<<<(unsigned)((total + 255) / 256), B256, 0, stream>>>(
        pl_ids, pl_table, ar_ids, ar_table, ttab, xA);
    track_gemm<<<NTRc / 16, B256, 0, stream>>>(track_x, Wtr_b, btr, ttab, xA);
  }

  // 3) degrees -> inv_deg (in place)
  zero_f32<<<2048, B256, 0, stream>>>(deg, (long long)NTOT);
  degree_kernel<<<(unsigned)((ETOT + 255) / 256), B256, 0, stream>>>(src_pt, dst_pt, src_ta, dst_ta, deg);
  invdeg_kernel<<<(NTOT + 255) / 256, B256, 0, stream>>>(deg);

  const long long aggN = (long long)NTOT * HD;
  const unsigned scatterBlocks = (unsigned)((ETOT + 7) / 8);

  // 4) SAGE layer 0: xA -> xB
  zero_f32<<<8192, B256, 0, stream>>>(agg, aggN);
  scatter_kernel<<<scatterBlocks, B256, 0, stream>>>(xA, agg, src_pt, dst_pt, src_ta, dst_ta);
  sage_kernel<<<NTOT / 16, B256, 0, stream>>>(xA, agg, deg, Wl0_b, Wr0_b, bl0, xB);

  // 5) SAGE layer 1: xB -> xA
  zero_f32<<<8192, B256, 0, stream>>>(agg, aggN);
  scatter_kernel<<<scatterBlocks, B256, 0, stream>>>(xB, agg, src_pt, dst_pt, src_ta, dst_ta);
  sage_kernel<<<NTOT / 16, B256, 0, stream>>>(xB, agg, deg, Wl1_b, Wr1_b, bl1, xA);

  // 6) prediction head
  predh_kernel<<<Lc / 16, B256, 0, stream>>>(xA, lab_row, lab_col, Wp1_b, bp1, hbuf);
  logits_kernel<<<(Lc + 7) / 8, B256, 0, stream>>>(hbuf, Wp2, bp2, out);
}
